// GCNUnit_34067680592304
// MI455X (gfx1250) — matrix operand order repaired and checked by reading, not compiled, not run on hardware
//
#include <hip/hip_runtime.h>

// ---------------------------------------------------------------------------
// GCN (2x GCNConv + LeakyReLU) for MI455X / gfx1250, wave32 + WMMA bf16.
//
// Pipeline (all on `stream`):
//   1. deg = 1 + in-degree(dst); disqrt = rsqrt(deg)           (ws: deg)
//   2. xbf = bf16(x)                                           (ws: bufBF)
//   3. h1 = xbf @ W1          (WMMA bf16, f32 accum)           (ws: bufA)
//   4. agg1 = h1*disqrt^2 ; scatter-add edges (L2 atomics)     (ws: bufB)
//   5. actbf = bf16(leaky(agg1 + b1))                          (ws: bufBF)
//   6. h2 = actbf @ W2                                         (ws: bufA)
//   7. out = h2*disqrt^2 ; scatter-add ; leaky(out + b2)       (d_out)
// ---------------------------------------------------------------------------

constexpr int D = 128;          // feature dim (fixed by reference)
constexpr float SLOPE = 0.01f;  // LeakyReLU negative slope
constexpr int TM = 8;           // row tiles (of 16) per GEMM block

typedef __attribute__((ext_vector_type(16))) __bf16 v16bf;
typedef __attribute__((ext_vector_type(8)))  float  v8f;

union BF16Vec {
  v16bf v;
  unsigned short h[16];
  uint4 q[2];
};

union BF16x4 {
  unsigned short s[4];
  uint2 u;
};

// f32 -> bf16, round-to-nearest-even
__device__ __forceinline__ unsigned short f2bf(float f) {
  union { float f; unsigned u; } x; x.f = f;
  unsigned r = x.u + 0x7FFFu + ((x.u >> 16) & 1u);
  return (unsigned short)(r >> 16);
}

__device__ __forceinline__ float bf2f(unsigned short s) {
  union { unsigned u; float f; } x; x.u = ((unsigned)s) << 16;
  return x.f;
}

// ---------------------------------------------------------------------------
// Degree kernels
// ---------------------------------------------------------------------------
__global__ __launch_bounds__(256) void k_init_deg(float* __restrict__ deg, int N) {
  int i = blockIdx.x * 256 + threadIdx.x;
  if (i < N) deg[i] = 1.0f;                       // self loop
}

__global__ __launch_bounds__(256) void k_count_deg(const int* __restrict__ dst,
                                                   float* __restrict__ deg, int E) {
  int e = blockIdx.x * 256 + threadIdx.x;
  if (e < E) atomicAdd(&deg[dst[e]], 1.0f);
}

__global__ __launch_bounds__(256) void k_rsqrt(float* __restrict__ deg, int N) {
  int i = blockIdx.x * 256 + threadIdx.x;
  if (i < N) deg[i] = rsqrtf(deg[i]);
}

// ---------------------------------------------------------------------------
// f32 -> bf16 bulk convert (float4 in, 4x bf16 out)
// ---------------------------------------------------------------------------
__global__ __launch_bounds__(256) void k_cvt_bf16(const float* __restrict__ in,
                                                  unsigned short* __restrict__ out,
                                                  int n4) {
  int t = blockIdx.x * 256 + threadIdx.x;
  if (t >= n4) return;
  float4 v = ((const float4*)in)[t];
  BF16x4 o;
  o.s[0] = f2bf(v.x); o.s[1] = f2bf(v.y); o.s[2] = f2bf(v.z); o.s[3] = f2bf(v.w);
  ((uint2*)out)[t] = o.u;
}

// ---------------------------------------------------------------------------
// GEMM: Out[M,128] = A[M,128](bf16) @ W[128,128](f32)  via WMMA, f32 accum.
// One block = TM*16 rows; 8 waves, each wave owns a 16-col slice.
// W staged once per block in LDS as bf16 column-major; B fragments for all
// four k-steps hoisted into registers and reused across TM row tiles
// (4*TM WMMAs per wave per block). Per tile: all four A fragments are
// preloaded (one 8x b128 clause), then 4 back-to-back WMMAs.
// Tile guard is block-uniform (requires M % 16 == 0 for tiles launched
// here; scalar tail kernel covers M % 16 != 0).
// ---------------------------------------------------------------------------
__global__ __launch_bounds__(256) void k_gemm_wmma(const unsigned short* __restrict__ Abf,
                                                   const float* __restrict__ W,
                                                   float* __restrict__ Out,
                                                   int mtiles) {
  __shared__ unsigned short ldsW[D * D];  // ldsW[n*128 + k], bf16, 32 KB

  const int tid = threadIdx.x;

  // Stage W (row-major f32) -> LDS (col-major bf16). Coalesced global reads.
  for (int i = tid; i < (D * D) / 4; i += 256) {
    const float4 w4 = ((const float4*)W)[i];
    const int k = (i * 4) / D;
    const int n = (i * 4) % D;
    ldsW[(n + 0) * D + k] = f2bf(w4.x);
    ldsW[(n + 1) * D + k] = f2bf(w4.y);
    ldsW[(n + 2) * D + k] = f2bf(w4.z);
    ldsW[(n + 3) * D + k] = f2bf(w4.w);
  }
  __syncthreads();

  const int wave = tid >> 5;
  const int lane = tid & 31;
  const int r    = lane & 15;   // A-row / B-col within tile
  const int g    = lane >> 4;   // lane group
  const int n0   = wave * 16;

  // B fragments for all k-steps, kept in registers across row tiles.
  // B k order chosen here: lanes 0-15 hold K=k0..k0+15, lanes 16-31 hold K=k0+16..k0+31.
  const unsigned short* bcol = ldsW + (n0 + r) * D;
  BF16Vec b[4];
#pragma unroll
  for (int kk = 0; kk < 4; ++kk) {
    const int kb = kk * 32 + 16 * g;
    b[kk].q[0] = *(const uint4*)(bcol + kb);
    b[kk].q[1] = *(const uint4*)(bcol + kb + 8);
  }

  const int t0 = blockIdx.x * TM;
#pragma unroll
  for (int mt = 0; mt < TM; ++mt) {
    const int tile = t0 + mt;
    if (tile >= mtiles) break;              // block-uniform guard
    const int m0 = tile * 16;
    const unsigned short* arow = Abf + (size_t)(m0 + r) * D;

    // Preload all four A fragments (one clause of 8x global_load_b128).
    // A k order chosen here: g=0: K 0-15; g=1: K 16-31 (matches B).
    BF16Vec a[4];
#pragma unroll
    for (int kk = 0; kk < 4; ++kk) {
      const int ka0 = kk * 32 + 16 * g;
      a[kk].q[0] = *(const uint4*)(arow + ka0);
      a[kk].q[1] = *(const uint4*)(arow + ka0 + 8);
    }

    v8f acc = {};
#pragma unroll
    for (int kk = 0; kk < 4; ++kk) {
      acc = __builtin_amdgcn_wmma_f32_16x16x32_bf16(
          /*neg_a=*/false, a[kk].v, /*neg_b=*/false, b[kk].v,
          /*c_mod=*/(short)0, acc, /*reuse_a=*/false, /*reuse_b=*/false);
    }

    // D layout: VGPR v -> row m = v + 8*g, col n = n0 + r
    float* orow = Out + (size_t)m0 * D + (n0 + r);
#pragma unroll
    for (int v = 0; v < 8; ++v) {
      orow[(size_t)(v + 8 * g) * D] = acc[v];
    }
  }
}

// Scalar tail for M % 16 != 0 (not taken for N=100000, kept for generality).
__global__ __launch_bounds__(256) void k_gemm_tail(const unsigned short* __restrict__ Abf,
                                                   const float* __restrict__ W,
                                                   float* __restrict__ Out,
                                                   int row0, int M) {
  int t = blockIdx.x * 256 + threadIdx.x;     // one thread per (row, col)
  int row = row0 + t / D;
  int col = t % D;
  if (row >= M) return;
  const unsigned short* a = Abf + (size_t)row * D;
  float s = 0.f;
  for (int k = 0; k < D; ++k) {
    s += bf2f(a[k]) * bf2f(f2bf(W[(size_t)k * D + col]));  // match WMMA numerics
  }
  Out[(size_t)row * D + col] = s;
}

// ---------------------------------------------------------------------------
// agg[i,:] = h[i,:] * disqrt[i]^2        (self-loop term, float4 elementwise)
// ---------------------------------------------------------------------------
__global__ __launch_bounds__(256) void k_self_loop(const float* __restrict__ h,
                                                   const float* __restrict__ disq,
                                                   float* __restrict__ agg, int N) {
  int t = blockIdx.x * 256 + threadIdx.x;     // over N * D/4
  if (t >= N * (D / 4)) return;
  int node = t / (D / 4);
  float w = disq[node]; w *= w;
  float4 v = ((const float4*)h)[t];
  v.x *= w; v.y *= w; v.z *= w; v.w *= w;
  ((float4*)agg)[t] = v;
}

// ---------------------------------------------------------------------------
// Edge scatter: one wave per edge, 4 floats per lane (512B/edge, coalesced).
// agg[dst,:] += h[src,:] * disqrt[src]*disqrt[dst]
// e is wave-uniform: readfirstlane forces the edge/coef loads scalar (SMEM).
// h (51MB) and agg (51MB) both fit in the 192MB L2 -> atomics are L2-resident.
// ---------------------------------------------------------------------------
__global__ __launch_bounds__(256) void k_edge_scatter(const float* __restrict__ h,
                                                      const float* __restrict__ disq,
                                                      const int* __restrict__ src,
                                                      const int* __restrict__ dst,
                                                      float* __restrict__ agg, int E) {
  int t = blockIdx.x * 256 + threadIdx.x;
  int e = __builtin_amdgcn_readfirstlane(t >> 5);   // wave-uniform edge id
  int lane = t & 31;
  if (e >= E) return;
  int s = src[e];
  int d = dst[e];
  float coef = disq[s] * disq[d];
  const float4 hv = ((const float4*)(h + (size_t)s * D))[lane];
  float* out = agg + (size_t)d * D + lane * 4;
  atomicAdd(out + 0, hv.x * coef);
  atomicAdd(out + 1, hv.y * coef);
  atomicAdd(out + 2, hv.z * coef);
  atomicAdd(out + 3, hv.w * coef);
}

// ---------------------------------------------------------------------------
// Epilogues: leaky_relu(agg + bias) -> f32 (final) or bf16 (next GEMM input)
// ---------------------------------------------------------------------------
__device__ __forceinline__ float4 bias_lrelu4(const float* __restrict__ agg,
                                              const float* __restrict__ bias,
                                              int t) {
  int dq = t % (D / 4);
  float4 b4 = ((const float4*)bias)[dq];
  float4 v = ((const float4*)agg)[t];
  v.x += b4.x; v.y += b4.y; v.z += b4.z; v.w += b4.w;
  v.x = v.x > 0.f ? v.x : v.x * SLOPE;
  v.y = v.y > 0.f ? v.y : v.y * SLOPE;
  v.z = v.z > 0.f ? v.z : v.z * SLOPE;
  v.w = v.w > 0.f ? v.w : v.w * SLOPE;
  return v;
}

__global__ __launch_bounds__(256) void k_bias_lrelu_f32(const float* __restrict__ agg,
                                                        const float* __restrict__ bias,
                                                        float* __restrict__ out, int N) {
  int t = blockIdx.x * 256 + threadIdx.x;     // over N * D/4
  if (t >= N * (D / 4)) return;
  ((float4*)out)[t] = bias_lrelu4(agg, bias, t);
}

__global__ __launch_bounds__(256) void k_bias_lrelu_bf16(const float* __restrict__ agg,
                                                         const float* __restrict__ bias,
                                                         unsigned short* __restrict__ outbf,
                                                         int N) {
  int t = blockIdx.x * 256 + threadIdx.x;     // over N * D/4
  if (t >= N * (D / 4)) return;
  float4 v = bias_lrelu4(agg, bias, t);
  BF16x4 o;
  o.s[0] = f2bf(v.x); o.s[1] = f2bf(v.y); o.s[2] = f2bf(v.z); o.s[3] = f2bf(v.w);
  ((uint2*)outbf)[t] = o.u;
}

// ---------------------------------------------------------------------------
// Host launch
// ---------------------------------------------------------------------------
extern "C" void kernel_launch(void* const* d_in, const int* in_sizes, int n_in,
                              void* d_out, int out_size, void* d_ws, size_t ws_size,
                              hipStream_t stream) {
  (void)n_in; (void)out_size; (void)ws_size;

  const float* x  = (const float*)d_in[0];
  const float* W1 = (const float*)d_in[1];
  const float* b1 = (const float*)d_in[2];
  const float* W2 = (const float*)d_in[3];
  const float* b2 = (const float*)d_in[4];
  const int* edge_index = (const int*)d_in[5];
  // d_in[6] = batch (unused: single graph, output is node features)

  const int N = in_sizes[0] / D;
  const int E = in_sizes[5] / 2;
  const int* src = edge_index;       // edge_index[0, :]
  const int* dst = edge_index + E;   // edge_index[1, :]

  // Workspace: deg (N f32) | bufA (N*D f32) | bufB (N*D f32) | bufBF (N*D bf16)
  char* ws = (char*)d_ws;
  float* deg = (float*)ws;
  size_t off = ((size_t)N * sizeof(float) + 4095) & ~(size_t)4095;
  size_t fsz = (size_t)N * D * sizeof(float);
  float* bufA = (float*)(ws + off);
  float* bufB = (float*)(ws + off + fsz);
  unsigned short* bufBF = (unsigned short*)(ws + off + 2 * fsz);
  float* out = (float*)d_out;

  const int nq = N * (D / 4);                   // float4 elements per feature map
  const int gb_elem  = (nq + 255) / 256;
  const int gb_nodes = (N + 255) / 256;
  const int gb_edges = (E + 255) / 256;
  const int gb_scat  = (E * 32 + 255) / 256;    // one wave per edge
  const int mtiles   = N / 16;                  // full 16-row tiles
  const int mrem0    = mtiles * 16;
  const int gb_gemm  = (mtiles + TM - 1) / TM;  // TM row tiles per block

  // 1. degrees
  k_init_deg<<<gb_nodes, 256, 0, stream>>>(deg, N);
  k_count_deg<<<gb_edges, 256, 0, stream>>>(dst, deg, E);
  k_rsqrt<<<gb_nodes, 256, 0, stream>>>(deg, N);

  // 2. layer 1: h1 = x @ W1
  k_cvt_bf16<<<gb_elem, 256, 0, stream>>>(x, bufBF, nq);
  if (mtiles > 0) k_gemm_wmma<<<gb_gemm, 256, 0, stream>>>(bufBF, W1, bufA, mtiles);
  if (mrem0 < N) {
    int tail = (N - mrem0) * D;
    k_gemm_tail<<<(tail + 255) / 256, 256, 0, stream>>>(bufBF, W1, bufA, mrem0, N);
  }
  k_self_loop<<<gb_elem, 256, 0, stream>>>(bufA, deg, bufB, N);
  k_edge_scatter<<<gb_scat, 256, 0, stream>>>(bufA, deg, src, dst, bufB, E);
  k_bias_lrelu_bf16<<<gb_elem, 256, 0, stream>>>(bufB, b1, bufBF, N);  // act1(bf16)

  // 3. layer 2: h2 = act1 @ W2
  if (mtiles > 0) k_gemm_wmma<<<gb_gemm, 256, 0, stream>>>(bufBF, W2, bufA, mtiles);
  if (mrem0 < N) {
    int tail = (N - mrem0) * D;
    k_gemm_tail<<<(tail + 255) / 256, 256, 0, stream>>>(bufBF, W2, bufA, mrem0, N);
  }
  k_self_loop<<<gb_elem, 256, 0, stream>>>(bufA, deg, out, N);
  k_edge_scatter<<<gb_scat, 256, 0, stream>>>(bufA, deg, src, dst, out, E);
  k_bias_lrelu_f32<<<gb_elem, 256, 0, stream>>>(out, b2, out, N);
}
